// HiCAP_54460185313467
// MI455X (gfx1250) — matrix operand order repaired and checked by reading, not compiled, not run on hardware
//
#include <hip/hip_runtime.h>
#include <hip/hip_bf16.h>
#include <math.h>

// ---------------------------------------------------------------------------
// CDNA5 (gfx1250) hierarchical GCN pooling. wave32. GEMMs: TDM-staged weights
// in LDS + v_wmma_f32_16x16x32_f16 (fp32 accumulate), fully unrolled K loop.
// ---------------------------------------------------------------------------

typedef _Float16 v16h  __attribute__((ext_vector_type(16)));
typedef _Float16 v8h   __attribute__((ext_vector_type(8)));
typedef float    v8f   __attribute__((ext_vector_type(8)));
typedef unsigned u32x4 __attribute__((ext_vector_type(4)));
typedef unsigned u32x8 __attribute__((ext_vector_type(8)));

#define USE_TDM 1

// ---------------- float order-preserving encode for atomic max -------------
__device__ __forceinline__ unsigned enc_f(float f) {
    unsigned u = __float_as_uint(f);
    return (u & 0x80000000u) ? ~u : (u | 0x80000000u);
}
__device__ __forceinline__ float dec_f(unsigned u) {
    unsigned b = (u & 0x80000000u) ? (u & 0x7fffffffu) : ~u;
    return __uint_as_float(b);
}

// ---------------- degree / normalization -----------------------------------
__global__ void k_deg_count(const int* __restrict__ dst, float* __restrict__ deg, int E) {
    int t = blockIdx.x * blockDim.x + threadIdx.x;
    if (t < E) atomicAdd(&deg[dst[t]], 1.0f);
}
__global__ void k_rsqrt(const float* __restrict__ deg, float* __restrict__ inv, int n) {
    int t = blockIdx.x * blockDim.x + threadIdx.x;
    if (t < n) inv[t] = rsqrtf(deg[t] + 1.0f);
}

// ---------------- WMMA GEMM: T = X[n,K] @ W[K,m] ---------------------------
// Block = 256 threads = 8 waves. blockIdx.x = column tile (16 cols of W),
// blockIdx.y*8 + waveId = row tile. The K x 16 f32 slab of W is DMA'd into
// LDS by the Tensor Data Mover once per block, transposed/converted to f16
// in LDS, then all 8 waves stream it through v_wmma_f32_16x16x32_f16.
//
// A frag (f16 16x32): lanes 0-15 row M=lane  K={0..7,16..23};
//                     lanes 16-31 row M=lane-16 K={8..15,24..31}.
// B frag (f16 32x16): lanes 0-15 col N=lane  K={0..7,16..23};
//                     lanes 16-31 col N=lane-16 K={8..15,24..31}.
template <int K>
__global__ void k_gemm_wmma(const float* __restrict__ X, const float* __restrict__ W,
                            float* __restrict__ T, int n, int m) {
    constexpr int KPAD = K + 8;                 // f16 row pad, keeps 16B alignment
    __shared__ __align__(16) unsigned char smem[K * 16 * 4 + 16 * KPAD * 2];
    float*    sW32 = (float*)smem;              // [K][16] f32 tile (LDS offset 0)
    _Float16* sBT  = (_Float16*)(smem + K * 64);// [16 cols][KPAD] f16 transposed

    const int wid   = threadIdx.x >> 5;
    const int lane  = threadIdx.x & 31;
    const int tileC = blockIdx.x;
    const int tileR = blockIdx.y * 8 + wid;
    const bool hi   = lane >= 16;
    const bool active = (tileR * 16) < n;

    // ---- stage W[:, tileC*16 : tileC*16+16] into LDS -----------------------
#if USE_TDM
    if (threadIdx.x < 32) {
        unsigned long long ga =
            (unsigned long long)(const void*)(W + (size_t)tileC * 16);
        u32x4 g0;
        g0[0] = 1u;                                        // count=1, user mode
        g0[1] = 0u;                                        // lds_addr = 0 (sW32)
        g0[2] = (unsigned)(ga & 0xFFFFFFFFu);              // global_addr lo
        g0[3] = (unsigned)((ga >> 32) & 0x01FFFFFFu) | (2u << 30); // hi | type=2
        u32x8 g1;
        g1[0] = (2u << 16);                                // data_size = 4B
        g1[1] = ((unsigned)m & 0xFFFFu) << 16;             // tensor_dim0 lo16
        g1[2] = ((unsigned)K & 0xFFFFu) << 16;             // dim0 hi=0 | dim1 lo16
        g1[3] = (16u << 16);                               // dim1 hi=0 | tile_dim0=16
        g1[4] = (unsigned)K;                               // tile_dim1=K, tile_dim2=0
        g1[5] = (unsigned)m;                               // tensor_dim0_stride lo32
        g1[6] = 0u;                                        // stride hi | dim1_stride
        g1[7] = 0u;
        asm volatile("tensor_load_to_lds %0, %1" : : "s"(g0), "s"(g1) : "memory");
        __builtin_amdgcn_s_wait_tensorcnt(0);
    }
    __syncthreads();
#else
    for (int idx = threadIdx.x; idx < K * 16; idx += blockDim.x) {
        int kk = idx >> 4, c = idx & 15;
        sW32[idx] = W[(size_t)kk * m + tileC * 16 + c];
    }
    __syncthreads();
#endif
    // transpose + f32->f16 convert inside LDS
    for (int idx = threadIdx.x; idx < K * 16; idx += blockDim.x) {
        int kk = idx >> 4, c = idx & 15;
        sBT[c * KPAD + kk] = (_Float16)sW32[idx];
    }
    __syncthreads();

    if (active) {
        const int mrow = tileR * 16 + (lane & 15);
        const _Float16* bcol = sBT + (lane & 15) * KPAD + (hi ? 8 : 0);
        v8f acc = {};
#pragma unroll
        for (int k0 = 0; k0 < K; k0 += 32) {
            // A fragment: two contiguous 8-float groups -> b128 loads
            const float* ap = X + (size_t)mrow * K + k0 + (hi ? 8 : 0);
            v16h a;
#pragma unroll
            for (int i = 0; i < 8; ++i) a[i]     = (_Float16)ap[i];
#pragma unroll
            for (int i = 0; i < 8; ++i) a[8 + i] = (_Float16)ap[16 + i];

            // B fragment: two aligned 16-byte LDS loads
            const _Float16* bp = bcol + k0;
            v8h blo = *(const v8h*)bp;
            v8h bhi = *(const v8h*)(bp + 16);
            v16h b;
#pragma unroll
            for (int i = 0; i < 8; ++i) { b[i] = blo[i]; b[8 + i] = bhi[i]; }

            acc = __builtin_amdgcn_wmma_f32_16x16x32_f16(
                false, a, false, b, (short)0, acc, false, false);
        }
        // C/D layout: VGPR r -> lanes 0-15: M=r; lanes 16-31: M=r+8
        const int orow = tileR * 16 + (hi ? 8 : 0);
        float* op = T + (size_t)(tileC * 16 + (lane & 15));
#pragma unroll
        for (int r = 0; r < 8; ++r) op[(size_t)(orow + r) * m] = acc[r];
    }
}

// ---------------- edge scatter: Agg[dst] += T[src] * inv[s]*inv[d] ---------
__global__ void k_edge_agg(const int* __restrict__ src, const int* __restrict__ dst,
                           const float* __restrict__ inv, const float* __restrict__ T,
                           float* __restrict__ Agg, int E, int width) {
    int wave = (blockIdx.x * blockDim.x + threadIdx.x) >> 5;
    int lane = threadIdx.x & 31;
    if (wave >= E) return;
    int s = src[wave], d = dst[wave];
    float coef = inv[s] * inv[d];
    const float* tp = T + (size_t)s * width;
    float* ap = Agg + (size_t)d * width;
    for (int f = lane; f < width; f += 32)
        atomicAdd(&ap[f], tp[f] * coef);
}

// ---------------- gcn finalize: O = relu(Agg + T*inv^2 + b) ----------------
__global__ void k_gcn_finalize(const float* __restrict__ Agg, const float* __restrict__ T,
                               const float* __restrict__ inv, const float* __restrict__ bias,
                               float* __restrict__ O, int n, int width) {
    size_t t = (size_t)blockIdx.x * blockDim.x + threadIdx.x;
    size_t total = (size_t)n * width;
    if (t >= total) return;
    int i = (int)(t / width);
    int f = (int)(t % width);
    float iv = inv[i];
    float v = Agg[t] + T[t] * iv * iv + bias[f];
    O[t] = fmaxf(v, 0.0f);
}

// ---------------- row softmax + argmax (wave per node, c=512) --------------
__global__ void k_softmax_argmax(float* __restrict__ S, int* __restrict__ cluster,
                                 int n, int c) {
    int wave = (blockIdx.x * blockDim.x + threadIdx.x) >> 5;
    int lane = threadIdx.x & 31;
    if (wave >= n) return;
    float* row = S + (size_t)wave * c;
    int per = c >> 5;
    float v[16];
    float best = -INFINITY;
    int   bidx = 0x7fffffff;
    for (int j = 0; j < per; ++j) {
        int idx = lane + j * 32;
        float val = row[idx];
        v[j] = val;
        if (val > best) { best = val; bidx = idx; }
    }
    for (int msk = 16; msk >= 1; msk >>= 1) {
        float ov = __shfl_xor(best, msk, 32);
        int   oi = __shfl_xor(bidx, msk, 32);
        if (ov > best || (ov == best && oi < bidx)) { best = ov; bidx = oi; }
    }
    float sum = 0.0f;
    for (int j = 0; j < per; ++j) { v[j] = expf(v[j] - best); sum += v[j]; }
    for (int msk = 16; msk >= 1; msk >>= 1) sum += __shfl_xor(sum, msk, 32);
    float rs = 1.0f / sum;
    for (int j = 0; j < per; ++j) row[lane + j * 32] = v[j] * rs;
    if (lane == 0) cluster[wave] = bidx;
}

// ---------------- score mat-vec: hs[i] = dot(x[i,:], Ws) -------------------
__global__ void k_score_mv(const float* __restrict__ x, const float* __restrict__ Wsv,
                           float* __restrict__ hs, int n, int f) {
    int wave = (blockIdx.x * blockDim.x + threadIdx.x) >> 5;
    int lane = threadIdx.x & 31;
    if (wave >= n) return;
    const float* xp = x + (size_t)wave * f;
    float s = 0.0f;
    for (int j = lane; j < f; j += 32) s += xp[j] * Wsv[j];
    for (int msk = 16; msk >= 1; msk >>= 1) s += __shfl_xor(s, msk, 32);
    if (lane == 0) hs[wave] = s;
}

// ---------------- intra-cluster degree + nonempty counts -------------------
__global__ void k_intra_deg(const int* __restrict__ src, const int* __restrict__ dst,
                            const int* __restrict__ cl, float* __restrict__ degs,
                            int* __restrict__ csum, int E) {
    int t = blockIdx.x * blockDim.x + threadIdx.x;
    if (t >= E) return;
    int s = src[t], d = dst[t];
    int cs = cl[s];
    if (cs == cl[d]) {
        atomicAdd(&degs[d], 1.0f);
        atomicAdd(&csum[cs], 1);
    }
}

// ---------------- score edge scatter (scalar) ------------------------------
__global__ void k_score_edge(const int* __restrict__ src, const int* __restrict__ dst,
                             const int* __restrict__ cl, const float* __restrict__ hs,
                             const float* __restrict__ invs, float* __restrict__ aggs, int E) {
    int t = blockIdx.x * blockDim.x + threadIdx.x;
    if (t >= E) return;
    int s = src[t], d = dst[t];
    if (cl[s] == cl[d]) atomicAdd(&aggs[d], hs[s] * invs[s] * invs[d]);
}

__global__ void k_score_finalize(const float* __restrict__ aggs, const float* __restrict__ hs,
                                 const float* __restrict__ invs, const float* __restrict__ bs,
                                 float* __restrict__ score, int n) {
    int t = blockIdx.x * blockDim.x + threadIdx.x;
    if (t >= n) return;
    float iv = invs[t];
    score[t] = tanhf(aggs[t] + hs[t] * iv * iv + bs[0]);
}

// ---------------- segment max / argmin-of-max ------------------------------
__global__ void k_seg_max(const float* __restrict__ score, const int* __restrict__ cl,
                          unsigned* __restrict__ segmax, int n) {
    int t = blockIdx.x * blockDim.x + threadIdx.x;
    if (t < n) atomicMax(&segmax[cl[t]], enc_f(score[t]));
}
__global__ void k_init_u32(unsigned* __restrict__ p, int n, unsigned val) {
    int t = blockIdx.x * blockDim.x + threadIdx.x;
    if (t < n) p[t] = val;
}
__global__ void k_seg_idx(const float* __restrict__ score, const int* __restrict__ cl,
                          const unsigned* __restrict__ segmax, unsigned* __restrict__ idxArr,
                          int n) {
    int t = blockIdx.x * blockDim.x + threadIdx.x;
    if (t >= n) return;
    int c = cl[t];
    if (enc_f(score[t]) >= segmax[c]) atomicMin(&idxArr[c], (unsigned)t);
}

// ---------------- pooling + new_batch --------------------------------------
__global__ void k_pool(const float* __restrict__ x, const int* __restrict__ batch,
                       const unsigned* __restrict__ segmax, const unsigned* __restrict__ idxArr,
                       const int* __restrict__ csum, float* __restrict__ new_x,
                       float* __restrict__ new_batch, int n, int f) {
    int c = blockIdx.x;
    int t = threadIdx.x;                       // blockDim.x == f
    bool ne = csum[c] > 0;
    float alpha = ne ? dec_f(segmax[c]) : 0.0f;
    unsigned si = idxArr[c];
    if (si > (unsigned)(n - 1)) si = (unsigned)(n - 1);
    new_x[(size_t)c * f + t] = x[(size_t)si * f + t] * alpha;
    if (t == 0) new_batch[c] = ne ? (float)batch[si] : 0.0f;
}

// ---------------- coarse adjacency (idempotent 1.0f stores) ----------------
__global__ void k_adj(const int* __restrict__ src, const int* __restrict__ dst,
                      const int* __restrict__ cl, const int* __restrict__ csum,
                      float* __restrict__ A, int E, int c) {
    int t = blockIdx.x * blockDim.x + threadIdx.x;
    if (t >= E) return;
    int ci = cl[src[t]], cj = cl[dst[t]];
    if (ci != cj && csum[ci] > 0 && csum[cj] > 0)
        A[(size_t)ci * c + cj] = 1.0f;
}

// ===========================================================================
extern "C" void kernel_launch(void* const* d_in, const int* in_sizes, int n_in,
                              void* d_out, int out_size, void* d_ws, size_t ws_size,
                              hipStream_t stream) {
    (void)n_in; (void)out_size; (void)ws_size;

    const float* x     = (const float*)d_in[0];
    const int*   eidx  = (const int*)  d_in[1];
    const int*   batch = (const int*)  d_in[2];
    const float* W1    = (const float*)d_in[3];
    const float* b1    = (const float*)d_in[4];
    const float* W2    = (const float*)d_in[5];
    const float* b2    = (const float*)d_in[6];
    const float* W3    = (const float*)d_in[7];
    const float* b3    = (const float*)d_in[8];
    const float* Wsv   = (const float*)d_in[9];
    const float* bs    = (const float*)d_in[10];

    const int N = in_sizes[2];            // 50000
    const int F = in_sizes[0] / N;        // 128
    const int H = in_sizes[4];            // 256
    const int C = in_sizes[8];            // 512
    const int E = in_sizes[1] / 2;        // 800000
    const int* src = eidx;
    const int* dst = eidx + E;

    // ---- workspace carve-out -------------------------------------------------
    char* ws = (char*)d_ws;
    size_t off = 0;
    auto alloc = [&](size_t bytes) -> void* {
        void* p = ws + off;
        off += (bytes + 255) & ~(size_t)255;
        return p;
    };
    float*    P0      = (float*)   alloc((size_t)N * C * 4);  // T1 / T3
    float*    P1      = (float*)   alloc((size_t)N * H * 4);  // T2 / h2
    float*    Agg     = (float*)   alloc((size_t)N * C * 4);
    float*    deg     = (float*)   alloc((size_t)N * 4);
    float*    inv     = (float*)   alloc((size_t)N * 4);
    float*    hs      = (float*)   alloc((size_t)N * 4);
    float*    degs    = (float*)   alloc((size_t)N * 4);
    float*    invs    = (float*)   alloc((size_t)N * 4);
    float*    aggs    = (float*)   alloc((size_t)N * 4);
    float*    score   = (float*)   alloc((size_t)N * 4);
    int*      cluster = (int*)     alloc((size_t)N * 4);
    unsigned* segmax  = (unsigned*)alloc((size_t)C * 4);
    unsigned* idxArr  = (unsigned*)alloc((size_t)C * 4);
    int*      csum    = (int*)     alloc((size_t)C * 4);

    // ---- output carve-out ----------------------------------------------------
    float* out      = (float*)d_out;
    float* newx_out = out;                                   // [C,F]
    float* adj_out  = out + (size_t)C * F;                   // [C,C]
    float* nb_out   = out + (size_t)C * F + (size_t)C * C;   // [C]
    float* S_out    = nb_out + C;                            // [N,C]

    const int TPB = 256;
    auto cdiv = [](long long a, long long b) { return (unsigned)((a + b - 1) / b); };

    // ---- degrees -------------------------------------------------------------
    hipMemsetAsync(deg, 0, (size_t)N * 4, stream);
    k_deg_count<<<cdiv(E, TPB), TPB, 0, stream>>>(dst, deg, E);
    k_rsqrt<<<cdiv(N, TPB), TPB, 0, stream>>>(deg, inv, N);

    const unsigned rowBlocks = cdiv(N / 16, 8);

    // ---- layer 1: h1 = relu(gcn(x, W1))  (F=128 -> H), in-place into P0 ------
    {
        dim3 grid(H / 16, rowBlocks);
        k_gemm_wmma<128><<<grid, TPB, 0, stream>>>(x, W1, P0, N, H);
        hipMemsetAsync(Agg, 0, (size_t)N * H * 4, stream);
        k_edge_agg<<<cdiv(E, 8), TPB, 0, stream>>>(src, dst, inv, P0, Agg, E, H);
        k_gcn_finalize<<<cdiv((long long)N * H, TPB), TPB, 0, stream>>>(Agg, P0, inv, b1, P0, N, H);
    }
    // ---- layer 2: h2 = relu(gcn(h1, W2)) (H=256 -> H), in-place into P1 ------
    {
        dim3 grid(H / 16, rowBlocks);
        k_gemm_wmma<256><<<grid, TPB, 0, stream>>>(P0, W2, P1, N, H);
        hipMemsetAsync(Agg, 0, (size_t)N * H * 4, stream);
        k_edge_agg<<<cdiv(E, 8), TPB, 0, stream>>>(src, dst, inv, P1, Agg, E, H);
        k_gcn_finalize<<<cdiv((long long)N * H, TPB), TPB, 0, stream>>>(Agg, P1, inv, b2, P1, N, H);
    }
    // ---- layer 3: h3 = relu(gcn(h2, W3)) (H=256 -> C), final into S_out ------
    {
        dim3 grid(C / 16, rowBlocks);
        k_gemm_wmma<256><<<grid, TPB, 0, stream>>>(P1, W3, P0, N, C);
        hipMemsetAsync(Agg, 0, (size_t)N * C * 4, stream);
        k_edge_agg<<<cdiv(E, 8), TPB, 0, stream>>>(src, dst, inv, P0, Agg, E, C);
        k_gcn_finalize<<<cdiv((long long)N * C, TPB), TPB, 0, stream>>>(Agg, P0, inv, b3, S_out, N, C);
    }

    // ---- softmax + argmax -> S, cluster --------------------------------------
    k_softmax_argmax<<<cdiv((long long)N * 32, TPB), TPB, 0, stream>>>(S_out, cluster, N, C);

    // ---- score layer on intra-cluster subgraph -------------------------------
    k_score_mv<<<cdiv((long long)N * 32, TPB), TPB, 0, stream>>>(x, Wsv, hs, N, F);
    hipMemsetAsync(degs, 0, (size_t)N * 4, stream);
    hipMemsetAsync(csum, 0, (size_t)C * 4, stream);
    hipMemsetAsync(aggs, 0, (size_t)N * 4, stream);
    k_intra_deg<<<cdiv(E, TPB), TPB, 0, stream>>>(src, dst, cluster, degs, csum, E);
    k_rsqrt<<<cdiv(N, TPB), TPB, 0, stream>>>(degs, invs, N);
    k_score_edge<<<cdiv(E, TPB), TPB, 0, stream>>>(src, dst, cluster, hs, invs, aggs, E);
    k_score_finalize<<<cdiv(N, TPB), TPB, 0, stream>>>(aggs, hs, invs, bs, score, N);

    // ---- per-cluster max score + argmax node ---------------------------------
    hipMemsetAsync(segmax, 0, (size_t)C * 4, stream);   // enc 0 == -inf
    k_seg_max<<<cdiv(N, TPB), TPB, 0, stream>>>(score, cluster, segmax, N);
    k_init_u32<<<cdiv(C, TPB), TPB, 0, stream>>>(idxArr, C, (unsigned)N);
    k_seg_idx<<<cdiv(N, TPB), TPB, 0, stream>>>(score, cluster, segmax, idxArr, N);

    // ---- pooling + coarse adjacency ------------------------------------------
    k_pool<<<C, F, 0, stream>>>(x, batch, segmax, idxArr, csum, newx_out, nb_out, N, F);
    hipMemsetAsync(adj_out, 0, (size_t)C * C * 4, stream);
    k_adj<<<cdiv(E, TPB), TPB, 0, stream>>>(src, dst, cluster, csum, adj_out, E, C);
}
